// LongFormerContentModel_25116968747095
// MI455X (gfx1250) — matrix operand order repaired and checked by reading, not compiled, not run on hardware
//
#include <hip/hip_runtime.h>

// ---------------------------------------------------------------------------
// Longformer-base forward for MI455X (gfx1250, wave32, WMMA).
// All GEMMs run on v_wmma_f32_16x16x32_bf16 (bf16 in, f32 acc).
// Weights are transposed at conversion time so every WMMA operand fragment is
// loaded with contiguous global_load_b128s; the GEMM hot loop is a ping-pong
// software pipeline (two fragment register sets, no copies) so loads for the
// next 32-K step overlap the current step's WMMAs.
// Attention is flash-style with Longformer band/global chunk skipping.
// ---------------------------------------------------------------------------

typedef __attribute__((ext_vector_type(16))) __bf16 v16bf;
typedef __attribute__((ext_vector_type(8)))  float  v8f;

#define WONE 256          // one-sided window
#define NEGBIG (-1e9f)

__device__ __forceinline__ v8f wmma_bf16(v16bf a, v16bf b, v8f c) {
  // (neg_a, A, neg_b, B, c_mod, C, reuse_a, reuse_b)
  return __builtin_amdgcn_wmma_f32_16x16x32_bf16(false, a, false, b, (short)0, c,
                                                 false, false);
}

union Frag32B {
  uint4 u[2];
  v16bf v;
};

// Combine two 16-byte spans into one 32-byte fragment register block.
__device__ __forceinline__ v16bf load_pair16(const __bf16* p0,
                                             const __bf16* p1) {
  Frag32B f;
  f.u[0] = *(const uint4*)p0;
  f.u[1] = *(const uint4*)p1;
  return f.v;
}

__device__ __forceinline__ v16bf zero_frag() {
  Frag32B f;
  f.u[0] = make_uint4(0, 0, 0, 0);
  f.u[1] = make_uint4(0, 0, 0, 0);
  return f.v;
}

// ---- A fragment (16x32 bf16, ISA 7.12.2) -------------------------------
// lane m=l&15 holds row m; half=l>>4 selects K base; per lane the fragment is
// two contiguous 16B spans: k = hl*8 + [0,8) and k = 16 + hl*8 + [0,8).
__device__ __forceinline__ v16bf load_a_fast(const __bf16* __restrict__ base,
                                             size_t rowBase, size_t colBase,
                                             size_t ld) {
  const int l = threadIdx.x & 31, hl = l >> 4, m = l & 15;
  const __bf16* p0 = base + (rowBase + m) * ld + colBase + hl * 8;
  return load_pair16(p0, p0 + 16);
}

// Guarded (cold path, partial M): the guard is uniform across the fragment,
// so it's a single exec-masked region with the same two b128 loads.
__device__ __forceinline__ v16bf load_a_guard(const __bf16* __restrict__ base,
                                              size_t rowBase, size_t colBase,
                                              size_t ld, int mValid) {
  const int l = threadIdx.x & 31, hl = l >> 4, m = l & 15;
  if (m < mValid) {
    const __bf16* p0 = base + (rowBase + m) * ld + colBase + hl * 8;
    return load_pair16(p0, p0 + 16);
  }
  return zero_frag();
}

// A fragment from a 16x32 LDS tile (row-major, stride 32): two ds_load_b128.
__device__ __forceinline__ v16bf load_a_lds(const __bf16* Pl) {
  const int l = threadIdx.x & 31, hl = l >> 4, m = l & 15;
  const __bf16* p0 = Pl + m * 32 + hl * 8;
  return load_pair16(p0, p0 + 16);
}

// ---- B fragment (32x16 bf16) from a source where K is contiguous --------
// element (k,n) = src[(rowStart+n)*ld + colStart + k]; per lane n the 16
// elements are two 16B spans: k = hl*8 + [0,8) and k = 16 + hl*8 + [0,8).
__device__ __forceinline__ v16bf load_b_contig(const __bf16* __restrict__ src,
                                               size_t rowStart, size_t colStart,
                                               size_t ld) {
  const int l = threadIdx.x & 31, hl = l >> 4, n = l & 15;
  const __bf16* p0 = src + (rowStart + n) * ld + colStart + hl * 8;
  return load_pair16(p0, p0 + 16);
}

// Guarded variant: lanes with (n >= nValid) supply zeros (cold path, N<16).
__device__ __forceinline__ v16bf load_b_contig_g(const __bf16* __restrict__ src,
                                                 size_t rowStart,
                                                 size_t colStart, size_t ld,
                                                 int nValid) {
  const int l = threadIdx.x & 31, hl = l >> 4, n = l & 15;
  if (n < nValid) {
    const __bf16* p0 = src + (rowStart + n) * ld + colStart + hl * 8;
    return load_pair16(p0, p0 + 16);
  }
  return zero_frag();
}

// ---------------------------------------------------------------------------
// WMMA GEMM: out = act(alpha * (A[M,K] x W + bias)), W given TRANSPOSED as
// Wt[N,K]. One wave per 16x64 output tile; the hot loop is a copy-free
// ping-pong pipeline over two fragment sets (requires K % 64 == 0).
// act: 0=none, 1=gelu, 2=tanh. transB: write bf16 output transposed ([N,M]).
// ---------------------------------------------------------------------------
__global__ __launch_bounds__(32) void gemm_bf16_kernel(
    const __bf16* __restrict__ A, const __bf16* __restrict__ Wt,
    const float* __restrict__ bias, float* __restrict__ outF,
    __bf16* __restrict__ outB, int M, int N, int K, float alpha, int act,
    int transB) {
  const int l = threadIdx.x & 31, hl = l >> 4, n = l & 15;
  const int mBase = blockIdx.x * 16;
  const int nBase0 = blockIdx.y * 64;
  const int mValid = (M - mBase) < 16 ? (M - mBase) : 16;

  v8f acc[4] = {};
  if (mValid == 16 && nBase0 + 64 <= N && (K & 63) == 0) {
    // hot path: pure b128 traffic; ping-pong over two fragment sets
    const size_t nb0 = (size_t)nBase0, nb1 = nb0 + 16, nb2 = nb0 + 32,
                 nb3 = nb0 + 48;
    const size_t mB = (size_t)mBase, sK = (size_t)K;
    v16bf a0 = load_a_fast(A, mB, 0, sK);
    v16bf b00 = load_b_contig(Wt, nb0, 0, sK);
    v16bf b01 = load_b_contig(Wt, nb1, 0, sK);
    v16bf b02 = load_b_contig(Wt, nb2, 0, sK);
    v16bf b03 = load_b_contig(Wt, nb3, 0, sK);
    int k = 0;
    for (;;) {
      // stage 1 loads (k+32) issued before stage 0 WMMAs consume set 0
      const size_t k1 = (size_t)(k + 32);
      v16bf a1 = load_a_fast(A, mB, k1, sK);
      v16bf b10 = load_b_contig(Wt, nb0, k1, sK);
      v16bf b11 = load_b_contig(Wt, nb1, k1, sK);
      v16bf b12 = load_b_contig(Wt, nb2, k1, sK);
      v16bf b13 = load_b_contig(Wt, nb3, k1, sK);
      __builtin_prefetch(Wt + (nb0 + n) * sK + k + 64, 0, 1);  // warm L2 ahead
      acc[0] = wmma_bf16(a0, b00, acc[0]);
      acc[1] = wmma_bf16(a0, b01, acc[1]);
      acc[2] = wmma_bf16(a0, b02, acc[2]);
      acc[3] = wmma_bf16(a0, b03, acc[3]);
      k += 64;
      if (k >= K) {
        acc[0] = wmma_bf16(a1, b10, acc[0]);
        acc[1] = wmma_bf16(a1, b11, acc[1]);
        acc[2] = wmma_bf16(a1, b12, acc[2]);
        acc[3] = wmma_bf16(a1, b13, acc[3]);
        break;
      }
      const size_t k0 = (size_t)k;
      a0 = load_a_fast(A, mB, k0, sK);
      b00 = load_b_contig(Wt, nb0, k0, sK);
      b01 = load_b_contig(Wt, nb1, k0, sK);
      b02 = load_b_contig(Wt, nb2, k0, sK);
      b03 = load_b_contig(Wt, nb3, k0, sK);
      acc[0] = wmma_bf16(a1, b10, acc[0]);
      acc[1] = wmma_bf16(a1, b11, acc[1]);
      acc[2] = wmma_bf16(a1, b12, acc[2]);
      acc[3] = wmma_bf16(a1, b13, acc[3]);
    }
  } else {
    // cold path: head GEMMs (M=2 / N=2)
    for (int k = 0; k < K; k += 32) {
      v16bf a = load_a_guard(A, (size_t)mBase, (size_t)k, (size_t)K, mValid);
#pragma unroll
      for (int t = 0; t < 4; ++t) {
        int nb = nBase0 + t * 16;
        if (nb < N) {
          int nValid = (N - nb) < 16 ? (N - nb) : 16;
          v16bf bw =
              load_b_contig_g(Wt, (size_t)nb, (size_t)k, (size_t)K, nValid);
          acc[t] = wmma_bf16(a, bw, acc[t]);
        }
      }
    }
  }
#pragma unroll
  for (int t = 0; t < 4; ++t) {
    int nn = nBase0 + t * 16 + n;
    if (nn >= N) continue;
    float bval = bias ? bias[nn] : 0.0f;
#pragma unroll
    for (int r = 0; r < 8; ++r) {
      int mm = mBase + hl * 8 + r;
      if (mm >= M) continue;
      float v = (acc[t][r] + bval) * alpha;
      if (act == 1)
        v = 0.5f * v * (1.0f + erff(v * 0.70710678118654752f));
      else if (act == 2)
        v = tanhf(v);
      if (outF) outF[(size_t)mm * N + nn] = v;
      if (outB) {
        size_t idx = transB ? ((size_t)nn * M + mm) : ((size_t)mm * N + nn);
        outB[idx] = (__bf16)v;
      }
    }
  }
}

// ---------------------------------------------------------------------------
// Flash attention, one wave per (query-16-tile, head, batch).
// variant 0: local (band | global-key) mask, chunk skipping.
// variant 1: global rows (dense over key_ok).
// Q pre-scaled by 1/sqrt(DH). Q/K: [B*S, D] (head h = cols h*64..h*64+63).
// V is TRANSPOSED: Vt[D, B*S].
// ---------------------------------------------------------------------------
__global__ __launch_bounds__(32) void attn_kernel(
    const __bf16* __restrict__ Q, const __bf16* __restrict__ Kt,
    const __bf16* __restrict__ Vt, const int* __restrict__ key_ok,
    const int* __restrict__ is_g, float* __restrict__ outp, int S, int Dm,
    int Mtot, int variant) {
  const int l = threadIdx.x & 31;
  const int hl = l >> 4, n = l & 15;
  const int qBase = blockIdx.x * 16;
  const int cOff = blockIdx.y * 64;
  const size_t rowB = (size_t)blockIdx.z * S;

  v16bf aq0 = load_a_fast(Q, rowB + qBase, (size_t)cOff, (size_t)Dm);
  v16bf aq1 = load_a_fast(Q, rowB + qBase, (size_t)cOff + 32, (size_t)Dm);

  v8f acc0 = {}, acc1 = {}, acc2 = {}, acc3 = {};
  float mrow[8], lrow[8];
#pragma unroll
  for (int r = 0; r < 8; ++r) {
    mrow[r] = -3.0e38f;
    lrow[r] = 0.0f;
  }

  __shared__ __bf16 Pl[16 * 32];

  const int nChunks = S / 32;
  for (int c = 0; c < nChunks; ++c) {
    const int kBase = c * 32;
    if (variant == 0) {
      // Longformer structure: skip chunks outside the band with no global keys.
      bool band = (kBase <= qBase + 15 + WONE) && (kBase + 31 >= qBase - WONE);
      bool anyg = __any(is_g[rowB + kBase + l] != 0);
      if (!band && !anyg) continue;
    }
    // scores for 32 keys: issue all 4 fragment loads, then 4 WMMAs so the
    // compiler can stagger s_wait_loadcnt instead of fully serializing.
    v8f s0 = {}, s1 = {};
    {
      v16bf b00 = load_b_contig(Kt, rowB + kBase, (size_t)cOff, (size_t)Dm);
      v16bf b01 = load_b_contig(Kt, rowB + kBase + 16, (size_t)cOff, (size_t)Dm);
      v16bf b10 = load_b_contig(Kt, rowB + kBase, (size_t)cOff + 32, (size_t)Dm);
      v16bf b11 =
          load_b_contig(Kt, rowB + kBase + 16, (size_t)cOff + 32, (size_t)Dm);
      s0 = wmma_bf16(aq0, b00, s0);
      s1 = wmma_bf16(aq0, b01, s1);
      s0 = wmma_bf16(aq1, b10, s0);
      s1 = wmma_bf16(aq1, b11, s1);
    }
    // mask (this lane's two key columns are fixed across r)
    const int kj0 = kBase + n, kj1 = kBase + 16 + n;
    const int ok0 = key_ok[rowB + kj0], ok1 = key_ok[rowB + kj1];
    const int g0 = is_g[rowB + kj0], g1 = is_g[rowB + kj1];
#pragma unroll
    for (int r = 0; r < 8; ++r) {
      int qi = qBase + hl * 8 + r;
      bool a0, a1;
      if (variant == 0) {
        int d0 = qi - kj0; if (d0 < 0) d0 = -d0;
        int d1 = qi - kj1; if (d1 < 0) d1 = -d1;
        a0 = ok0 && (g0 || d0 <= WONE);
        a1 = ok1 && (g1 || d1 <= WONE);
      } else {
        a0 = ok0 != 0;
        a1 = ok1 != 0;
      }
      if (!a0) s0[r] = NEGBIG;
      if (!a1) s1[r] = NEGBIG;
    }
    // online softmax: row M=hl*8+r lives in VGPR r across the 16 lanes of a
    // half-wave -> width-16 shuffle reductions; rescale factor maps 1:1 onto
    // the PV accumulator layout.
#pragma unroll
    for (int r = 0; r < 8; ++r) {
      float t = fmaxf(s0[r], s1[r]);
      t = fmaxf(t, __shfl_xor(t, 1, 16));
      t = fmaxf(t, __shfl_xor(t, 2, 16));
      t = fmaxf(t, __shfl_xor(t, 4, 16));
      t = fmaxf(t, __shfl_xor(t, 8, 16));
      float nm = fmaxf(mrow[r], t);
      float sc = __expf(mrow[r] - nm);
      mrow[r] = nm;
      float p0 = __expf(s0[r] - nm);
      float p1 = __expf(s1[r] - nm);
      float rs = p0 + p1;
      rs += __shfl_xor(rs, 1, 16);
      rs += __shfl_xor(rs, 2, 16);
      rs += __shfl_xor(rs, 4, 16);
      rs += __shfl_xor(rs, 8, 16);
      lrow[r] = lrow[r] * sc + rs;
      int row = hl * 8 + r;
      Pl[row * 32 + n] = (__bf16)p0;
      Pl[row * 32 + 16 + n] = (__bf16)p1;
      acc0[r] *= sc;
      acc1[r] *= sc;
      acc2[r] *= sc;
      acc3[r] *= sc;
    }
    // drain LDS stores before cross-lane reads (CDNA5 split DS counter)
    asm volatile("s_wait_dscnt 0" ::: "memory");
    v16bf aP = load_a_lds(Pl);
    {
      // Vt[D, Mtot]: B(k,n) = Vt[(cOff+t*16+n)*Mtot + rowB+kBase+k] -- contiguous
      v16bf bv0 = load_b_contig(Vt, (size_t)cOff + 0, rowB + kBase, (size_t)Mtot);
      v16bf bv1 = load_b_contig(Vt, (size_t)cOff + 16, rowB + kBase, (size_t)Mtot);
      v16bf bv2 = load_b_contig(Vt, (size_t)cOff + 32, rowB + kBase, (size_t)Mtot);
      v16bf bv3 = load_b_contig(Vt, (size_t)cOff + 48, rowB + kBase, (size_t)Mtot);
      acc0 = wmma_bf16(aP, bv0, acc0);
      acc1 = wmma_bf16(aP, bv1, acc1);
      acc2 = wmma_bf16(aP, bv2, acc2);
      acc3 = wmma_bf16(aP, bv3, acc3);
    }
  }
#pragma unroll
  for (int r = 0; r < 8; ++r) {
    float inv = lrow[r] > 0.0f ? 1.0f / lrow[r] : 0.0f;
    size_t base = (rowB + qBase + hl * 8 + r) * (size_t)Dm + cOff;
    outp[base + 0 + n] = acc0[r] * inv;
    outp[base + 16 + n] = acc1[r] * inv;
    outp[base + 32 + n] = acc2[r] * inv;
    outp[base + 48 + n] = acc3[r] * inv;
  }
}

// ---------------------------------------------------------------------------
// Support kernels
// ---------------------------------------------------------------------------

// fp32 [L,K,N] -> bf16 transposed [L,N,K], LDS-tiled for coalescing.
__global__ __launch_bounds__(256) void cvtT_kernel(const float* __restrict__ src,
                                                   __bf16* __restrict__ dst,
                                                   int K, int N) {
  __shared__ float tile[32][33];
  const int lidx = blockIdx.z;
  const int k0 = blockIdx.y * 32, n0 = blockIdx.x * 32;
  const float* s = src + (size_t)lidx * K * N;
  __bf16* d = dst + (size_t)lidx * K * N;
  const int tx = threadIdx.x & 31, ty = threadIdx.x >> 5;  // 32x8
  for (int i = ty; i < 32; i += 8) {
    int k = k0 + i, n = n0 + tx;
    tile[i][tx] = (k < K && n < N) ? s[(size_t)k * N + n] : 0.0f;
  }
  __syncthreads();
  for (int i = ty; i < 32; i += 8) {
    int n = n0 + i, k = k0 + tx;
    if (n < N && k < K) d[(size_t)n * K + k] = (__bf16)tile[tx][i];
  }
}

__global__ void embed_kernel(const int* __restrict__ ids,
                             const float* __restrict__ we,
                             const float* __restrict__ pe,
                             const float* __restrict__ te,
                             float* __restrict__ outp, int S, int D) {
  int row = blockIdx.x;  // b*S + s
  int s = row % S;
  int id = ids[row];
  for (int d = threadIdx.x; d < D; d += blockDim.x)
    outp[(size_t)row * D + d] =
        we[(size_t)id * D + d] + pe[(size_t)(s + 2) * D + d] + te[d];
}

// LayerNorm over last dim with optional residual; writes f32 and/or bf16.
__global__ __launch_bounds__(256) void add_ln_kernel(
    const float* __restrict__ x, const float* __restrict__ res,
    const float* __restrict__ g, const float* __restrict__ bta,
    float* __restrict__ outF, __bf16* __restrict__ outB, int D) {
  int row = blockIdx.x;
  const float* xr = x + (size_t)row * D;
  const float* rr = res ? res + (size_t)row * D : nullptr;
  float local[4];
  int cnt = 0;
  float sum = 0.0f;
  for (int d = threadIdx.x, i = 0; d < D; d += blockDim.x, ++i) {
    float v = xr[d] + (rr ? rr[d] : 0.0f);
    local[i] = v;
    sum += v;
    cnt = i + 1;
  }
  __shared__ float sred[8];
  int wid = threadIdx.x >> 5, lid = threadIdx.x & 31;
  for (int o = 16; o; o >>= 1) sum += __shfl_xor(sum, o, 32);
  if (lid == 0) sred[wid] = sum;
  __syncthreads();
  if (wid == 0) {
    float t = (lid < 8) ? sred[lid] : 0.0f;
    for (int o = 4; o; o >>= 1) t += __shfl_xor(t, o, 32);
    if (lid == 0) sred[0] = t;
  }
  __syncthreads();
  float mean = sred[0] / D;
  __syncthreads();
  float vs = 0.0f;
  for (int i = 0; i < cnt; ++i) {
    float dlt = local[i] - mean;
    vs += dlt * dlt;
  }
  for (int o = 16; o; o >>= 1) vs += __shfl_xor(vs, o, 32);
  if (lid == 0) sred[wid] = vs;
  __syncthreads();
  if (wid == 0) {
    float t = (lid < 8) ? sred[lid] : 0.0f;
    for (int o = 4; o; o >>= 1) t += __shfl_xor(t, o, 32);
    if (lid == 0) sred[0] = t;
  }
  __syncthreads();
  float rstd = rsqrtf(sred[0] / D + 1e-5f);
  for (int d = threadIdx.x, i = 0; d < D; d += blockDim.x, ++i) {
    float v = (local[i] - mean) * rstd * g[d] + bta[d];
    size_t idx = (size_t)row * D + d;
    if (outF) outF[idx] = v;
    if (outB) outB[idx] = (__bf16)v;
  }
}

// out[row,:] = bf16(is_g[row] ? out_g : out_loc); one block per row
__global__ void merge_kernel(const float* __restrict__ oloc,
                             const float* __restrict__ og,
                             const int* __restrict__ isg,
                             __bf16* __restrict__ outp, int D) {
  size_t row = blockIdx.x;
  const float* src = isg[row] ? (og + row * D) : (oloc + row * D);
  for (int d = threadIdx.x; d < D; d += blockDim.x)
    outp[row * D + d] = (__bf16)src[d];
}

// out[b,:] = bf16(h[b, 0, :])
__global__ void gather_cls_kernel(const float* __restrict__ h,
                                  __bf16* __restrict__ outp, int S, int D) {
  int b = blockIdx.x;
  for (int d = threadIdx.x; d < D; d += blockDim.x)
    outp[(size_t)b * D + d] = (__bf16)h[(size_t)b * S * D + d];
}

// ---------------------------------------------------------------------------
// Orchestration
// ---------------------------------------------------------------------------
extern "C" void kernel_launch(void* const* d_in, const int* in_sizes, int n_in,
                              void* d_out, int out_size, void* d_ws,
                              size_t ws_size, hipStream_t stream) {
  (void)in_sizes; (void)n_in; (void)out_size; (void)ws_size;
  const int B = 2, S = 1024, D = 768, FF = 3072, L = 12, NCLS = 2;
  const float SCALE = 0.125f;  // 1/sqrt(64)
  const int M = B * S;

  const int* input_ids = (const int*)d_in[0];
  const int* amask = (const int*)d_in[1];
  const int* gmask = (const int*)d_in[2];
  const float* word_emb = (const float*)d_in[3];
  const float* pos_emb = (const float*)d_in[4];
  const float* type_emb = (const float*)d_in[5];
  const float* eln_s = (const float*)d_in[6];
  const float* eln_b = (const float*)d_in[7];
  const float* Wq = (const float*)d_in[8];   const float* bq = (const float*)d_in[9];
  const float* Wk = (const float*)d_in[10];  const float* bk = (const float*)d_in[11];
  const float* Wv = (const float*)d_in[12];  const float* bv = (const float*)d_in[13];
  const float* Wqg = (const float*)d_in[14]; const float* bqg = (const float*)d_in[15];
  const float* Wkg = (const float*)d_in[16]; const float* bkg = (const float*)d_in[17];
  const float* Wvg = (const float*)d_in[18]; const float* bvg = (const float*)d_in[19];
  const float* Wo = (const float*)d_in[20];  const float* bo = (const float*)d_in[21];
  const float* ln1s = (const float*)d_in[22]; const float* ln1b = (const float*)d_in[23];
  const float* Wi = (const float*)d_in[24];  const float* bi = (const float*)d_in[25];
  const float* Wo2 = (const float*)d_in[26]; const float* bo2 = (const float*)d_in[27];
  const float* ln2s = (const float*)d_in[28]; const float* ln2b = (const float*)d_in[29];
  const float* poolW = (const float*)d_in[30]; const float* poolb = (const float*)d_in[31];
  const float* denseW = (const float*)d_in[32]; const float* denseb = (const float*)d_in[33];
  const float* clsW = (const float*)d_in[34]; const float* clsb = (const float*)d_in[35];

  // ---- carve workspace ----
  size_t off = 0;
  auto alloc = [&](size_t bytes) -> void* {
    off = (off + 255) & ~(size_t)255;
    void* p = (char*)d_ws + off;
    off += bytes;
    return p;
  };
  const size_t wDD = (size_t)L * D * D;
  const size_t wDF = (size_t)L * D * FF;
  __bf16* wqT = (__bf16*)alloc(wDD * 2);
  __bf16* wkT = (__bf16*)alloc(wDD * 2);
  __bf16* wvT = (__bf16*)alloc(wDD * 2);
  __bf16* wqgT = (__bf16*)alloc(wDD * 2);
  __bf16* wkgT = (__bf16*)alloc(wDD * 2);
  __bf16* wvgT = (__bf16*)alloc(wDD * 2);
  __bf16* woT = (__bf16*)alloc(wDD * 2);
  __bf16* wiT = (__bf16*)alloc(wDF * 2);
  __bf16* wo2T = (__bf16*)alloc(wDF * 2);
  __bf16* poolWT = (__bf16*)alloc((size_t)D * D * 2);
  __bf16* denseWT = (__bf16*)alloc((size_t)D * D * 2);
  __bf16* clsWT = (__bf16*)alloc((size_t)D * NCLS * 2);

  const size_t actF = (size_t)M * D;  // elements
  float* embSum = (float*)alloc(actF * 4);
  float* haF = (float*)alloc(actF * 4);
  float* htF = (float*)alloc(actF * 4);
  __bf16* hB = (__bf16*)alloc(actF * 2);
  __bf16* qB = (__bf16*)alloc(actF * 2);
  __bf16* kB = (__bf16*)alloc(actF * 2);
  __bf16* vTB = (__bf16*)alloc(actF * 2);   // transposed [D, M]
  __bf16* qgB = (__bf16*)alloc(actF * 2);
  __bf16* kgB = (__bf16*)alloc(actF * 2);
  __bf16* vgTB = (__bf16*)alloc(actF * 2);  // transposed [D, M]
  float* outLoc = (float*)alloc(actF * 4);
  float* outG = (float*)alloc(actF * 4);
  __bf16* attnB = (__bf16*)alloc(actF * 2);
  float* oF = (float*)alloc(actF * 4);
  __bf16* ffB = (__bf16*)alloc((size_t)M * FF * 2);
  float* ff2F = (float*)alloc(actF * 4);
  __bf16* clsInB = (__bf16*)alloc((size_t)B * D * 2);
  __bf16* pooledB = (__bf16*)alloc((size_t)B * D * 2);
  __bf16* xB = (__bf16*)alloc((size_t)B * D * 2);

  auto cvtT = [&](const float* src, __bf16* dst, int K, int N, int Lcnt) {
    dim3 g((N + 31) / 32, (K + 31) / 32, Lcnt);
    cvtT_kernel<<<g, 256, 0, stream>>>(src, dst, K, N);
  };
  auto gemm = [&](const __bf16* A, const __bf16* Wt, const float* bias,
                  float* oFp, __bf16* oBp, int Mg, int Ng, int Kg, float alpha,
                  int act, int transB) {
    dim3 g((Mg + 15) / 16, (Ng + 63) / 64);
    gemm_bf16_kernel<<<g, 32, 0, stream>>>(A, Wt, bias, oFp, oBp, Mg, Ng, Kg,
                                           alpha, act, transB);
  };

  // ---- weight conversion + transpose (deterministic every call) ----
  cvtT(Wq, wqT, D, D, L);   cvtT(Wk, wkT, D, D, L);   cvtT(Wv, wvT, D, D, L);
  cvtT(Wqg, wqgT, D, D, L); cvtT(Wkg, wkgT, D, D, L); cvtT(Wvg, wvgT, D, D, L);
  cvtT(Wo, woT, D, D, L);
  cvtT(Wi, wiT, D, FF, L);
  cvtT(Wo2, wo2T, FF, D, L);
  cvtT(poolW, poolWT, D, D, 1);
  cvtT(denseW, denseWT, D, D, 1);
  cvtT(clsW, clsWT, D, NCLS, 1);

  // ---- embeddings + LN ----
  embed_kernel<<<M, 256, 0, stream>>>(input_ids, word_emb, pos_emb, type_emb,
                                      embSum, S, D);
  add_ln_kernel<<<M, 256, 0, stream>>>(embSum, nullptr, eln_s, eln_b, haF, hB, D);

  // ---- transformer layers ----
  dim3 ag(S / 16, 12 /*H*/, B);
  for (int l = 0; l < L; ++l) {
    const size_t wo_ = (size_t)l * D * D;
    const size_t b_ = (size_t)l * D;
    gemm(hB, wqT + wo_, bq + b_, nullptr, qB, M, D, D, SCALE, 0, 0);
    gemm(hB, wkT + wo_, bk + b_, nullptr, kB, M, D, D, 1.0f, 0, 0);
    gemm(hB, wvT + wo_, bv + b_, nullptr, vTB, M, D, D, 1.0f, 0, 1);
    gemm(hB, wqgT + wo_, bqg + b_, nullptr, qgB, M, D, D, SCALE, 0, 0);
    gemm(hB, wkgT + wo_, bkg + b_, nullptr, kgB, M, D, D, 1.0f, 0, 0);
    gemm(hB, wvgT + wo_, bvg + b_, nullptr, vgTB, M, D, D, 1.0f, 0, 1);

    attn_kernel<<<ag, 32, 0, stream>>>(qB, kB, vTB, amask, gmask, outLoc, S, D,
                                       M, 0);
    attn_kernel<<<ag, 32, 0, stream>>>(qgB, kgB, vgTB, amask, gmask, outG, S, D,
                                       M, 1);
    merge_kernel<<<M, 256, 0, stream>>>(outLoc, outG, gmask, attnB, D);

    gemm(attnB, woT + wo_, bo + b_, oF, nullptr, M, D, D, 1.0f, 0, 0);
    add_ln_kernel<<<M, 256, 0, stream>>>(oF, haF, ln1s + b_, ln1b + b_, htF, hB,
                                         D);

    gemm(hB, wiT + (size_t)l * D * FF, bi + (size_t)l * FF, nullptr, ffB, M, FF,
         D, 1.0f, 1 /*gelu*/, 0);
    gemm(ffB, wo2T + (size_t)l * FF * D, bo2 + b_, ff2F, nullptr, M, D, FF, 1.0f,
         0, 0);
    add_ln_kernel<<<M, 256, 0, stream>>>(ff2F, htF, ln2s + b_, ln2b + b_, haF,
                                         hB, D);
  }

  // ---- pooler + classifier head ----
  gather_cls_kernel<<<B, 256, 0, stream>>>(haF, clsInB, S, D);
  gemm(clsInB, poolWT, poolb, nullptr, pooledB, B, D, D, 1.0f, 2 /*tanh*/, 0);
  gemm(pooledB, denseWT, denseb, nullptr, xB, B, D, D, 1.0f, 0, 0);
  gemm(xB, clsWT, clsb, (float*)d_out, nullptr, B, NCLS, D, 1.0f, 0, 0);
}
